// ImageEncoderViT_91345364451518
// MI455X (gfx1250) — matrix operand order repaired and checked by reading, not compiled, not run on hardware
//
#include <hip/hip_runtime.h>
#include <hip/hip_bf16.h>
#include <math.h>

// ---------------------------------------------------------------------------
// SAM ViT encoder for MI455X (gfx1250, wave32, WMMA bf16 path).
// GEMMs: double-buffered LDS pipeline, A-tile staged by the Tensor Data Mover
// (tensor_load_to_lds + s_wait_tensorcnt) when available, B-tile transposed
// through VGPRs so B-fragments are contiguous ds_load_b128.
// ---------------------------------------------------------------------------

typedef __bf16 bf16;
typedef __attribute__((ext_vector_type(16))) __bf16 v16bf;
typedef __attribute__((ext_vector_type(8)))  __bf16 v8bf;
typedef __attribute__((ext_vector_type(8)))  float  v8f;
typedef unsigned int u32x4 __attribute__((ext_vector_type(4)));
typedef int          i32x4 __attribute__((ext_vector_type(4)));
typedef int          i32x8 __attribute__((ext_vector_type(8)));

#define D_EMB   768
#define D_QKV   2304
#define N_HEAD  12
#define HDIM    64
#define N_TOK   2048            // 2 images * 32*32 tokens
#define ATT_SCALE 0.125f        // 64^-0.5

#if __has_builtin(__builtin_amdgcn_tensor_load_to_lds) && \
    __has_builtin(__builtin_amdgcn_s_wait_tensorcnt)
#define USE_TDM 1
#else
#define USE_TDM 0
#endif

static __device__ __forceinline__ v8f wmma_bf16(v16bf a, v16bf b, v8f c) {
  return __builtin_amdgcn_wmma_f32_16x16x32_bf16(false, a, false, b, (short)0, c,
                                                 false, false);
}
static __device__ __forceinline__ v16bf cat8(v8bf lo, v8bf hi) {
  v16bf r;
#pragma unroll
  for (int i = 0; i < 8; ++i) { r[i] = lo[i]; r[i + 8] = hi[i]; }
  return r;
}

#if USE_TDM
// LDS byte offset of a __shared__ object (AS3 pointers are 32-bit LDS offsets).
static __device__ __forceinline__ unsigned lds_off(const void* p) {
  return (unsigned)(unsigned long long)(__attribute__((address_space(3))) const void*)p;
}
// Issue one 2D TDM tile load: tile (tileW x tileH) 2-byte elements from a
// row-major tensor with row stride `strideElems`, zero-filling rows >= tensorH.
// D# bitfields per CDNA5 ISA ch.8 (group0 128b, group1 256b, groups 2/3 zero).
static __device__ __forceinline__ void tdm_load_2d(
    const void* gptr, unsigned ldsByteAddr,
    unsigned tileW, unsigned tileH,
    unsigned tensorW, unsigned tensorH,
    unsigned long long strideElems)
{
  unsigned long long ga = (unsigned long long)gptr;
  u32x4 g0;
  g0[0] = 1u;                                      // count=1, user descriptor
  g0[1] = ldsByteAddr;                             // lds_addr
  g0[2] = (unsigned)(ga & 0xffffffffu);            // global_addr[31:0]
  g0[3] = (unsigned)((ga >> 32) & 0x01ffffffu)     // global_addr[56:32]
        | (2u << 30);                              // type = 2 (image)
  i32x8 g1;
  g1[0] = (int)(1u << 16);                         // data_size=1 (2B), mask=0
  g1[1] = (int)((tensorW & 0xffffu) << 16);        // tensor_dim0[15:0] @bit48
  g1[2] = (int)((tensorW >> 16) & 0xffffu)         // tensor_dim0[31:16]
        | (int)((tensorH & 0xffffu) << 16);        // tensor_dim1[15:0]
  g1[3] = (int)((tensorH >> 16) & 0xffffu)         // tensor_dim1[31:16]
        | (int)((tileW & 0xffffu) << 16);          // tile_dim0
  g1[4] = (int)(tileH & 0xffffu);                  // tile_dim1 (tile_dim2=0)
  g1[5] = (int)(strideElems & 0xffffffffull);      // tensor_dim0_stride[31:0]
  g1[6] = (int)((strideElems >> 32) & 0xffffull);  // stride[47:32], dim1_stride lo=0
  g1[7] = 0;
  i32x4 gz; gz[0] = gz[1] = gz[2] = gz[3] = 0;
#if __has_include(<hip/amd_detail/amd_gfx1250_TDM.h>)
  i32x8 g4; 
#pragma unroll
  for (int i = 0; i < 8; ++i) g4[i] = 0;
  __builtin_amdgcn_tensor_load_to_lds(g0, g1, gz, gz, g4, 0);   // clang-23 arity
#else
  __builtin_amdgcn_tensor_load_to_lds(g0, g1, gz, gz, 0);       // ROCm 7.2 arity
#endif
}
#endif  // USE_TDM

// ---------------------------------------------------------------------------
// Generic bf16 GEMM: C[M,N] = A[M,K]*B[K,N] (+bias)(+gelu)(+resid), out f32/bf16
// 128x128 block tile, 8 waves (2x4), each wave 64x32 (4x2 WMMA tiles), BK=32.
// Double-buffered LDS, one barrier per K-step; A tile via TDM when available.
// ---------------------------------------------------------------------------
__global__ __launch_bounds__(256) void k_gemm(
    const bf16* __restrict__ A, const bf16* __restrict__ B,
    const float* __restrict__ bias, const float* __restrict__ resid,
    float* __restrict__ Cf, bf16* __restrict__ Cb,
    int M, int N, int K, int dogelu)
{
  __shared__ __align__(16) bf16 As[2][128][32];   // 16 KB ping-pong
  __shared__ __align__(16) bf16 BsT[2][128][48];  // 24 KB ping-pong, transposed
  const int tid  = threadIdx.x;
  const int lane = tid & 31, wave = tid >> 5;
  const int half = lane >> 4, l15 = lane & 15;
  const int wm = wave >> 2, wn = wave & 3;
  const int tileM = blockIdx.x * 128, tileN = blockIdx.y * 128;

  v8f acc[4][2];
#pragma unroll
  for (int i = 0; i < 4; ++i)
#pragma unroll
    for (int j = 0; j < 2; ++j)
#pragma unroll
      for (int r = 0; r < 8; ++r) acc[i][j][r] = 0.f;

  // ---- prologue: stage k0 = 0 into buffer 0 ----
#if USE_TDM
  if (wave == 0)
    tdm_load_2d(A + (size_t)tileM * K, lds_off(&As[0][0][0]),
                32, 128, 32, (unsigned)(M - tileM), (unsigned long long)K);
#else
#pragma unroll
  for (int i = 0; i < 2; ++i) {
    int v = i * 256 + tid;
    int row = v >> 2, kk = (v & 3) << 3;
    int gr = tileM + row;
    v8bf val;
#pragma unroll
    for (int j = 0; j < 8; ++j) val[j] = (bf16)0.0f;
    if (gr < M) val = *(const v8bf*)(A + (size_t)gr * K + kk);
    *(v8bf*)(&As[0][row][kk]) = val;
  }
#endif
#pragma unroll
  for (int i = 0; i < 2; ++i) {
    int v = i * 256 + tid;
    int kk = v >> 4, nn = (v & 15) << 3;
    v8bf val = *(const v8bf*)(B + (size_t)kk * N + tileN + nn);
#pragma unroll
    for (int j = 0; j < 8; ++j) BsT[0][nn + j][kk] = val[j];
  }

  int p = 0;
  for (int k0 = 0; k0 < K; k0 += 32) {
#if USE_TDM
    if (wave == 0) __builtin_amdgcn_s_wait_tensorcnt(0);  // A[p] DMA complete
#endif
    __syncthreads();                                      // publish buffer p
    const bool next = (k0 + 32) < K;

    // ---- start staging k0+32 into buffer p^1 (overlaps with WMMA below) ----
#if !USE_TDM
    v8bf aregs[2];
#endif
    v8bf bregs[2];
    if (next) {
#if USE_TDM
      if (wave == 0)
        tdm_load_2d(A + (size_t)tileM * K + (k0 + 32), lds_off(&As[p ^ 1][0][0]),
                    32, 128, 32, (unsigned)(M - tileM), (unsigned long long)K);
#else
#pragma unroll
      for (int i = 0; i < 2; ++i) {
        int v = i * 256 + tid;
        int row = v >> 2, kk = (v & 3) << 3;
        int gr = tileM + row;
#pragma unroll
        for (int j = 0; j < 8; ++j) aregs[i][j] = (bf16)0.0f;
        if (gr < M) aregs[i] = *(const v8bf*)(A + (size_t)gr * K + k0 + 32 + kk);
      }
#endif
#pragma unroll
      for (int i = 0; i < 2; ++i) {
        int v = i * 256 + tid;
        int kk = v >> 4, nn = (v & 15) << 3;
        bregs[i] = *(const v8bf*)(B + (size_t)(k0 + 32 + kk) * N + tileN + nn);
      }
      if (k0 + 64 < K)   // global_prefetch_b8 one tile ahead
        __builtin_prefetch(B + (size_t)(k0 + 64 + (tid >> 3)) * N + tileN, 0, 1);
    }

    // ---- compute buffer p: 8 WMMAs per wave ----
    const int koff = half * 8;
    v16bf bfrag[2];
#pragma unroll
    for (int ns = 0; ns < 2; ++ns) {
      int col = wn * 32 + ns * 16 + l15;
      bfrag[ns] = cat8(*(const v8bf*)(&BsT[p][col][koff]),
                       *(const v8bf*)(&BsT[p][col][koff + 16]));
    }
#pragma unroll
    for (int ms = 0; ms < 4; ++ms) {
      int row = wm * 64 + ms * 16 + l15;
      v16bf a = cat8(*(const v8bf*)(&As[p][row][koff]),
                     *(const v8bf*)(&As[p][row][koff + 16]));
#pragma unroll
      for (int ns = 0; ns < 2; ++ns)
        acc[ms][ns] = wmma_bf16(a, bfrag[ns], acc[ms][ns]);
    }

    // ---- finish staging into p^1 ----
    if (next) {
#if !USE_TDM
#pragma unroll
      for (int i = 0; i < 2; ++i) {
        int v = i * 256 + tid;
        int row = v >> 2, kk = (v & 3) << 3;
        *(v8bf*)(&As[p ^ 1][row][kk]) = aregs[i];
      }
#endif
#pragma unroll
      for (int i = 0; i < 2; ++i) {
        int v = i * 256 + tid;
        int kk = v >> 4, nn = (v & 15) << 3;
#pragma unroll
        for (int j = 0; j < 8; ++j) BsT[p ^ 1][nn + j][kk] = bregs[i][j];
      }
    }
    p ^= 1;
  }

  const int rowBase = tileM + wm * 64;
  const int colBase = tileN + wn * 32;
#pragma unroll
  for (int ms = 0; ms < 4; ++ms)
#pragma unroll
    for (int ns = 0; ns < 2; ++ns)
#pragma unroll
      for (int r = 0; r < 8; ++r) {
        int row = rowBase + ms * 16 + r + half * 8;   // C layout: vgpr r, lane half
        int col = colBase + ns * 16 + l15;
        if (row >= M) continue;
        float v = acc[ms][ns][r];
        if (bias) v += bias[col];
        if (dogelu) v = 0.5f * v * (1.f + erff(v * 0.70710678f));
        if (resid) v += resid[(size_t)row * N + col];
        size_t o = (size_t)row * N + col;
        if (Cf) Cf[o] = v;
        if (Cb) Cb[o] = (bf16)v;
      }
}

// ---------------------------------------------------------------------------
// Fused flash-style attention. 4 waves/block, each wave owns one 16-query tile
// of one (group g, head) pair. Groups are windows (L=196) or images (L=1024).
// relh/relw: precomputed decomposed rel-pos dot products, added to scores.
// Writes un-partitioned token-layout output directly (skips pad queries).
// ---------------------------------------------------------------------------
__global__ __launch_bounds__(128) void k_attn(
    const bf16* __restrict__ qkv, const float* __restrict__ relh,
    const float* __restrict__ relw, bf16* __restrict__ outb,
    int G, int L, int Hw, int Ww, int windowed)
{
  __shared__ __align__(16) bf16 sP[4][16][40];   // softmax tile restage (A-frag)
  __shared__ __align__(16) bf16 sV[4][32][72];   // V rows stage
  const int lane = threadIdx.x & 31;
  const int w    = threadIdx.x >> 5;
  const int half = lane >> 4, l15 = lane & 15;
  const int head = blockIdx.y;
  const int g    = blockIdx.z;
  const int qt   = blockIdx.x * 4 + w;
  const int qbase = qt * 16;
  if (qbase >= L) return;
  const size_t gbase = (size_t)g * L;

  // Q as two 16x32 A fragments (dims 0..31, 32..63)
  int qr = qbase + l15; if (qr >= L) qr = L - 1;
  const bf16* qp = qkv + (gbase + qr) * D_QKV + head * HDIM;
  const int koffA = half * 8;
  v16bf aq[2];
#pragma unroll
  for (int t = 0; t < 2; ++t)
    aq[t] = cat8(*(const v8bf*)(qp + t * 32 + koffA),
                 *(const v8bf*)(qp + t * 32 + koffA + 16));

  float rm[8], rl[8];
  v8f o[4];
#pragma unroll
  for (int r = 0; r < 8; ++r) { rm[r] = -1e30f; rl[r] = 0.f; }
#pragma unroll
  for (int d = 0; d < 4; ++d)
#pragma unroll
    for (int r = 0; r < 8; ++r) o[d][r] = 0.f;

  const size_t rbase = ((size_t)(g * N_HEAD + head)) * L;

  for (int kb = 0; kb < L; kb += 32) {
    // stage V tile 32 keys x dims 0..31 into LDS (coalesced rows)
    {
      int key = kb + lane; if (key >= L) key = L - 1;
      const bf16* vp = qkv + (gbase + key) * D_QKV + 2 * D_EMB + head * HDIM;
#pragma unroll
      for (int j = 0; j < 4; ++j)
        *(v8bf*)(&sV[w][lane][j * 8]) = *(const v8bf*)(vp + j * 8);
    }
    // scores S = Q K^T for 32 keys (two 16-col WMMA tiles)
    float vs[2][8];
#pragma unroll
    for (int nt = 0; nt < 2; ++nt) {
      int key = kb + nt * 16 + l15;
      int keyc = key < L ? key : L - 1;
      const bf16* kp = qkv + (gbase + keyc) * D_QKV + D_EMB + head * HDIM;
      v8f s;
#pragma unroll
      for (int r = 0; r < 8; ++r) s[r] = 0.f;
#pragma unroll
      for (int t = 0; t < 2; ++t) {
        v16bf bk = cat8(*(const v8bf*)(kp + t * 32 + koffA),
                        *(const v8bf*)(kp + t * 32 + koffA + 16));  // K^T B-frag, table order
        s = wmma_bf16(aq[t], bk, s);
      }
#pragma unroll
      for (int r = 0; r < 8; ++r) {
        int q = qbase + r + half * 8;
        float v;
        if (q < L && key < L)
          v = s[r] * ATT_SCALE
            + relh[(rbase + q) * Hw + key / Ww]
            + relw[(rbase + q) * Ww + key % Ww];
        else
          v = -1e30f;
        vs[nt][r] = v;
      }
    }
    // online softmax per query row (rows live per half-wave, shfl width 16)
#pragma unroll
    for (int r = 0; r < 8; ++r) {
      float tmax = fmaxf(vs[0][r], vs[1][r]);
#pragma unroll
      for (int sft = 8; sft > 0; sft >>= 1)
        tmax = fmaxf(tmax, __shfl_xor(tmax, sft, 16));
      float mnew = fmaxf(rm[r], tmax);
      float fac  = __expf(rm[r] - mnew);
      float rsum = 0.f;
#pragma unroll
      for (int nt = 0; nt < 2; ++nt) {
        float p = __expf(vs[nt][r] - mnew);
        rsum += p;
        sP[w][r + half * 8][nt * 16 + l15] = (bf16)p;
      }
#pragma unroll
      for (int sft = 8; sft > 0; sft >>= 1)
        rsum += __shfl_xor(rsum, sft, 16);
      rl[r] = rl[r] * fac + rsum;
      rm[r] = mnew;
#pragma unroll
      for (int d = 0; d < 4; ++d) o[d][r] *= fac;
    }
    // O += P @ V  (P re-read from LDS in A-fragment layout)
    v16bf ap = cat8(*(const v8bf*)(&sP[w][l15][koffA]),
                    *(const v8bf*)(&sP[w][l15][koffA + 16]));
#pragma unroll
    for (int d = 0; d < 4; ++d) {
      v16bf bv;
#pragma unroll
      for (int j = 0; j < 16; ++j)
        bv[j] = sV[w][half * 8 + j + (j & 8)][d * 16 + l15];
      o[d] = wmma_bf16(ap, bv, o[d]);
    }
  }

  // write back (window un-partition fused here)
#pragma unroll
  for (int r = 0; r < 8; ++r) {
    int q = qbase + r + half * 8;
    if (q >= L) continue;
    float invl = 1.0f / rl[r];
    int tok;
    if (windowed) {
      int img = g / 9, wi = g % 9;
      int y = (wi / 3) * 14 + q / 14;
      int x = (wi % 3) * 14 + q % 14;
      if (y >= 32 || x >= 32) continue;
      tok = img * 1024 + y * 32 + x;
    } else {
      tok = g * L + q;
    }
#pragma unroll
    for (int d = 0; d < 4; ++d)
      outb[(size_t)tok * D_EMB + head * HDIM + d * 16 + l15] =
          (bf16)(o[d][r] * invl);
  }
}

// ---------------------------------------------------------------------------
// Decomposed rel-pos bias: out[g,h,q,kd] = dot64(q_vec, table[qc - kd + D - 1])
// ---------------------------------------------------------------------------
__global__ void k_relpos(const bf16* __restrict__ qkv, const float* __restrict__ tab,
                         float* __restrict__ outp, int G, int L, int D, int Ww, int isW)
{
  size_t total = (size_t)G * N_HEAD * L * D;
  for (size_t idx = (size_t)blockIdx.x * blockDim.x + threadIdx.x; idx < total;
       idx += (size_t)gridDim.x * blockDim.x) {
    int kd = (int)(idx % D);
    size_t t = idx / D;
    int q = (int)(t % L); t /= L;
    int h = (int)(t % N_HEAD);
    int g = (int)(t / N_HEAD);
    int qc = isW ? (q % Ww) : (q / Ww);
    const float* tr = tab + (size_t)(qc - kd + D - 1) * HDIM;
    const bf16* qv = qkv + ((size_t)g * L + q) * D_QKV + h * HDIM;
    float s = 0.f;
#pragma unroll
    for (int d = 0; d < HDIM; ++d) s += (float)qv[d] * tr[d];
    outp[idx] = s;
  }
}

// ---------------------------------------------------------------------------
// Row LayerNorm (token-layout). Writes bf16 (for next GEMM) and/or f32.
// ---------------------------------------------------------------------------
__global__ __launch_bounds__(256) void k_ln(const float* __restrict__ x,
    const float* __restrict__ w, const float* __restrict__ b,
    bf16* __restrict__ ob, float* __restrict__ of, int C)
{
  __shared__ float red[256];
  const int row = blockIdx.x, tid = threadIdx.x;
  const float* xr = x + (size_t)row * C;
  float s = 0.f;
  for (int c = tid; c < C; c += 256) s += xr[c];
  red[tid] = s; __syncthreads();
  for (int o = 128; o > 0; o >>= 1) { if (tid < o) red[tid] += red[tid + o]; __syncthreads(); }
  float mean = red[0] / C; __syncthreads();
  float s2 = 0.f;
  for (int c = tid; c < C; c += 256) { float d = xr[c] - mean; s2 += d * d; }
  red[tid] = s2; __syncthreads();
  for (int o = 128; o > 0; o >>= 1) { if (tid < o) red[tid] += red[tid + o]; __syncthreads(); }
  float inv = rsqrtf(red[0] / C + 1e-6f);
  for (int c = tid; c < C; c += 256) {
    float v = (xr[c] - mean) * inv * w[c] + b[c];
    if (ob) ob[(size_t)row * C + c] = (bf16)v;
    if (of) of[(size_t)row * C + c] = v;
  }
}

// Final LN2d: normalize over C, write NCHW into d_out.
__global__ __launch_bounds__(256) void k_ln_nchw(const float* __restrict__ x,
    const float* __restrict__ w, const float* __restrict__ b,
    float* __restrict__ outp, int C)
{
  __shared__ float red[256];
  const int row = blockIdx.x, tid = threadIdx.x;
  const float* xr = x + (size_t)row * C;
  float s = 0.f;
  for (int c = tid; c < C; c += 256) s += xr[c];
  red[tid] = s; __syncthreads();
  for (int o = 128; o > 0; o >>= 1) { if (tid < o) red[tid] += red[tid + o]; __syncthreads(); }
  float mean = red[0] / C; __syncthreads();
  float s2 = 0.f;
  for (int c = tid; c < C; c += 256) { float d = xr[c] - mean; s2 += d * d; }
  red[tid] = s2; __syncthreads();
  for (int o = 128; o > 0; o >>= 1) { if (tid < o) red[tid] += red[tid + o]; __syncthreads(); }
  float inv = rsqrtf(red[0] / C + 1e-6f);
  const int img = row >> 10, p = row & 1023;
  for (int c = tid; c < C; c += 256) {
    float v = (xr[c] - mean) * inv * w[c] + b[c];
    outp[((size_t)(img * C + c) << 10) + p] = v;
  }
}

// ---------------------------------------------------------------------------
// Elementwise helpers
// ---------------------------------------------------------------------------
__global__ void k_cvt(const float* __restrict__ s, bf16* __restrict__ d, size_t n) {
  size_t i = (size_t)blockIdx.x * blockDim.x + threadIdx.x;
  if (i < n) d[i] = (bf16)s[i];
}
__global__ void k_cvt_T(const float* __restrict__ s, bf16* __restrict__ d, int R, int C) {
  int i = blockIdx.x * blockDim.x + threadIdx.x;
  if (i < R * C) { int r = i / C, c = i % C; d[(size_t)c * R + r] = (bf16)s[i]; }
}
__global__ void k_prep_conv2w(const float* __restrict__ w, bf16* __restrict__ d) {
  int i = blockIdx.x * blockDim.x + threadIdx.x;   // d[(t9*256+cin)][cout]
  if (i < 2304 * 256) {
    int kk = i / 256, n = i % 256;
    int t9 = kk / 256, cin = kk % 256;
    d[i] = (bf16)w[((size_t)(n * 256 + cin)) * 9 + t9];
  }
}
__global__ void k_im2col_patch(const float* __restrict__ x, bf16* __restrict__ d) {
  int i = blockIdx.x * blockDim.x + threadIdx.x;
  if (i < N_TOK * D_EMB) {
    int kk = i % D_EMB, tok = i / D_EMB;
    int ch = kk / 256, rem = kk % 256, iy = rem / 16, ix = rem % 16;
    int img = tok >> 10, p = tok & 1023, py = p >> 5, px = p & 31;
    d[i] = (bf16)x[(((size_t)(img * 3 + ch)) * 512 + py * 16 + iy) * 512 + px * 16 + ix];
  }
}
__global__ void k_addpos(float* __restrict__ t, const float* __restrict__ pos) {
  int i = blockIdx.x * blockDim.x + threadIdx.x;
  if (i < N_TOK * D_EMB) t[i] += pos[i % (1024 * D_EMB)];
}
__global__ void k_stage_win(const bf16* __restrict__ xn, bf16* __restrict__ d) {
  int i = blockIdx.x * blockDim.x + threadIdx.x;   // [18*196, 768] with zero pad
  if (i < 3528 * D_EMB) {
    int c = i % D_EMB, row = i / D_EMB;
    int g = row / 196, q = row % 196;
    int img = g / 9, wi = g % 9;
    int y = (wi / 3) * 14 + q / 14;
    int x = (wi % 3) * 14 + q % 14;
    d[i] = (y < 32 && x < 32)
         ? xn[((size_t)(img * 1024 + y * 32 + x)) * D_EMB + c] : (bf16)0.0f;
  }
}
__global__ void k_im2col3(const bf16* __restrict__ y1n, bf16* __restrict__ d) {
  int i = blockIdx.x * blockDim.x + threadIdx.x;   // [2048, 9*256]
  if (i < N_TOK * 2304) {
    int kk = i % 2304, row = i / 2304;
    int t9 = kk / 256, cin = kk % 256;
    int dy = t9 / 3 - 1, dx = t9 % 3 - 1;
    int img = row >> 10, p = row & 1023;
    int y = (p >> 5) + dy, x = (p & 31) + dx;
    d[i] = (y >= 0 && y < 32 && x >= 0 && x < 32)
         ? y1n[((size_t)(img * 1024 + y * 32 + x)) * 256 + cin] : (bf16)0.0f;
  }
}

// ---------------------------------------------------------------------------
// Host driver
// ---------------------------------------------------------------------------
#define GEMM(A_, B_, bias_, res_, Cf_, Cb_, M_, N_, K_, gelu_)                        \
  k_gemm<<<dim3(((M_) + 127) / 128, (N_) / 128), 256, 0, stream>>>(                   \
      A_, B_, bias_, res_, Cf_, Cb_, M_, N_, K_, gelu_)
#define EWB(n) dim3((unsigned)(((n) + 255) / 256)), dim3(256), 0, stream

extern "C" void kernel_launch(void* const* d_in, const int* in_sizes, int n_in,
                              void* d_out, int out_size, void* d_ws, size_t ws_size,
                              hipStream_t stream) {
  (void)in_sizes; (void)n_in; (void)out_size; (void)ws_size;
  const float* x      = (const float*)d_in[0];
  const float* patchw = (const float*)d_in[1];
  const float* patchb = (const float*)d_in[2];
  const float* pos    = (const float*)d_in[3];
  const float* n1w    = (const float*)d_in[4];
  const float* n1lw   = (const float*)d_in[5];
  const float* n1lb   = (const float*)d_in[6];
  const float* n2w    = (const float*)d_in[7];
  const float* n2lw   = (const float*)d_in[8];
  const float* n2lb   = (const float*)d_in[9];
  const float* blk[2][14];
  for (int b = 0; b < 2; ++b)
    for (int j = 0; j < 14; ++j) blk[b][j] = (const float*)d_in[10 + b * 14 + j];

  // ---- workspace carve ----
  char* base = (char*)d_ws; size_t off = 0;
  auto carve = [&](size_t bytes) -> char* {
    char* p = base + off; off = (off + bytes + 255) & ~(size_t)255; return p;
  };
  bf16* wp      = (bf16*)carve((size_t)768 * 768 * 2);
  bf16* qkvw[2], * projw[2], * fc1w[2], * fc2w[2];
  for (int b = 0; b < 2; ++b) {
    qkvw[b] = (bf16*)carve((size_t)768 * 2304 * 2);
    projw[b] = (bf16*)carve((size_t)768 * 768 * 2);
    fc1w[b] = (bf16*)carve((size_t)768 * 3072 * 2);
    fc2w[b] = (bf16*)carve((size_t)3072 * 768 * 2);
  }
  bf16* n1wb = (bf16*)carve((size_t)768 * 256 * 2);
  bf16* n2wb = (bf16*)carve((size_t)2304 * 256 * 2);

  bf16*  Xp      = (bf16*)carve((size_t)N_TOK * 768 * 2);
  float* tokA    = (float*)carve((size_t)N_TOK * 768 * 4);
  float* tokB    = (float*)carve((size_t)N_TOK * 768 * 4);
  bf16*  xn_tok  = (bf16*)carve((size_t)N_TOK * 768 * 2);
  bf16*  xnw     = (bf16*)carve((size_t)3528 * 768 * 2);
  bf16*  qkvbuf  = (bf16*)carve((size_t)3528 * 2304 * 2);
  float* relhB   = (float*)carve((size_t)786432 * 4);
  float* relwB   = (float*)carve((size_t)786432 * 4);
  bf16*  attnout = (bf16*)carve((size_t)N_TOK * 768 * 2);
  bf16*  mlp_h   = (bf16*)carve((size_t)N_TOK * 3072 * 2);
  bf16*  xfin    = (bf16*)carve((size_t)N_TOK * 768 * 2);
  float* y1      = (float*)carve((size_t)N_TOK * 256 * 4);
  bf16*  y1n     = (bf16*)carve((size_t)N_TOK * 256 * 2);
  bf16*  X2      = (bf16*)carve((size_t)N_TOK * 2304 * 2);
  float* y2      = (float*)carve((size_t)N_TOK * 256 * 4);

  // ---- weight prep (f32 -> bf16, transposed to [K,N] where needed) ----
  k_cvt_T<<<EWB(768 * 768)>>>(patchw, wp, 768, 768);
  for (int b = 0; b < 2; ++b) {
    k_cvt<<<EWB(768 * 2304)>>>(blk[b][2], qkvw[b], (size_t)768 * 2304);
    k_cvt<<<EWB(768 * 768)>>>(blk[b][4], projw[b], (size_t)768 * 768);
    k_cvt<<<EWB(768 * 3072)>>>(blk[b][10], fc1w[b], (size_t)768 * 3072);
    k_cvt<<<EWB(3072 * 768)>>>(blk[b][12], fc2w[b], (size_t)3072 * 768);
  }
  k_cvt_T<<<EWB(256 * 768)>>>(n1w, n1wb, 256, 768);
  k_prep_conv2w<<<EWB(2304 * 256)>>>(n2w, n2wb);

  // ---- patch embed: im2col + GEMM + pos-embed ----
  k_im2col_patch<<<EWB(N_TOK * 768)>>>(x, Xp);
  GEMM(Xp, wp, patchb, nullptr, tokA, nullptr, N_TOK, 768, 768, 0);
  k_addpos<<<EWB(N_TOK * 768)>>>(tokA, pos);

  // ---- block 0: windowed attention (18 windows of 196 tokens) ----
  k_ln<<<N_TOK, 256, 0, stream>>>(tokA, blk[0][0], blk[0][1], xn_tok, nullptr, 768);
  k_stage_win<<<EWB(3528 * 768)>>>(xn_tok, xnw);
  GEMM(xnw, qkvw[0], blk[0][3], nullptr, nullptr, qkvbuf, 3528, 2304, 768, 0);
  k_relpos<<<dim3(4096), dim3(256), 0, stream>>>(qkvbuf, blk[0][6], relhB, 18, 196, 14, 14, 0);
  k_relpos<<<dim3(4096), dim3(256), 0, stream>>>(qkvbuf, blk[0][7], relwB, 18, 196, 14, 14, 1);
  k_attn<<<dim3(4, N_HEAD, 18), 128, 0, stream>>>(qkvbuf, relhB, relwB, attnout,
                                                  18, 196, 14, 14, 1);
  GEMM(attnout, projw[0], blk[0][5], tokA, tokB, nullptr, N_TOK, 768, 768, 0);
  k_ln<<<N_TOK, 256, 0, stream>>>(tokB, blk[0][8], blk[0][9], xn_tok, nullptr, 768);
  GEMM(xn_tok, fc1w[0], blk[0][11], nullptr, nullptr, mlp_h, N_TOK, 3072, 768, 1);
  GEMM(mlp_h, fc2w[0], blk[0][13], tokB, tokA, nullptr, N_TOK, 768, 3072, 0);

  // ---- block 1: global attention (2 images of 1024 tokens) ----
  k_ln<<<N_TOK, 256, 0, stream>>>(tokA, blk[1][0], blk[1][1], xn_tok, nullptr, 768);
  GEMM(xn_tok, qkvw[1], blk[1][3], nullptr, nullptr, qkvbuf, N_TOK, 2304, 768, 0);
  k_relpos<<<dim3(4096), dim3(256), 0, stream>>>(qkvbuf, blk[1][6], relhB, 2, 1024, 32, 32, 0);
  k_relpos<<<dim3(4096), dim3(256), 0, stream>>>(qkvbuf, blk[1][7], relwB, 2, 1024, 32, 32, 1);
  k_attn<<<dim3(16, N_HEAD, 2), 128, 0, stream>>>(qkvbuf, relhB, relwB, attnout,
                                                  2, 1024, 32, 32, 0);
  GEMM(attnout, projw[1], blk[1][5], tokA, tokB, nullptr, N_TOK, 768, 768, 0);
  k_ln<<<N_TOK, 256, 0, stream>>>(tokB, blk[1][8], blk[1][9], xn_tok, nullptr, 768);
  GEMM(xn_tok, fc1w[1], blk[1][11], nullptr, nullptr, mlp_h, N_TOK, 3072, 768, 1);
  GEMM(mlp_h, fc2w[1], blk[1][13], tokB, tokA, nullptr, N_TOK, 768, 3072, 0);

  // ---- neck: 1x1 conv, LN2d, 3x3 conv, LN2d -> d_out NCHW ----
  k_cvt<<<EWB(N_TOK * 768)>>>(tokA, xfin, (size_t)N_TOK * 768);
  GEMM(xfin, n1wb, nullptr, nullptr, y1, nullptr, N_TOK, 256, 768, 0);
  k_ln<<<N_TOK, 256, 0, stream>>>(y1, n1lw, n1lb, y1n, nullptr, 256);
  k_im2col3<<<EWB(N_TOK * 2304)>>>(y1n, X2);
  GEMM(X2, n2wb, nullptr, nullptr, y2, nullptr, N_TOK, 256, 2304, 0);
  k_ln_nchw<<<N_TOK, 256, 0, stream>>>(y2, n2lw, n2lb, (float*)d_out, 256);
}
